// L_GCL_19129784336812
// MI455X (gfx1250) — matrix operand order repaired and checked by reading, not compiled, not run on hardware
//
#include <hip/hip_runtime.h>

#define F_IN   64
#define MSG    64
#define F_OUT  64
#define EDGE_F 16
#define KPAD_E 160   // 2*F_IN + 1 + EDGE_F = 145, padded to 5 k-steps of 32
#define KE_STEPS 5
#define KN   128
#define KN_STEPS 4

typedef __attribute__((ext_vector_type(16))) _Float16 v16h;
typedef __attribute__((ext_vector_type(8)))  _Float16 v8h;
typedef __attribute__((ext_vector_type(8)))  float    v8f;

union H2U { _Float16 h; unsigned short u; };

__device__ __forceinline__ unsigned pack2h(float a, float b) {
  H2U x, y; x.h = (_Float16)a; y.h = (_Float16)b;
  return (unsigned)x.u | ((unsigned)y.u << 16);
}
__device__ __forceinline__ float silu_f(float v)   { return v / (1.0f + __expf(-v)); }
__device__ __forceinline__ float ssign_f(float v)  { return v / (1.0f + fabsf(v)); }

// Pack weight matrix W (Kreal x Ncols, row-major) into WMMA B-fragment order:
// frag f = (ntile*ksteps + kstep); per frag, lane L holds column n = ntile*16 + L%16,
// K-base = kstep*32 + (L/16)*16, 8 packed f16 pairs (32 bytes) contiguous per lane.
__device__ void fill_wfrags(const float* __restrict__ W, int Kreal, int ksteps,
                            int Ncols, unsigned* dst, int tid, int nthr) {
  int ntiles = Ncols >> 4;
  int total  = ntiles * ksteps * 256;
  for (int i = tid; i < total; i += nthr) {
    int f = i >> 8, lane = (i >> 3) & 31, v = i & 7;
    int n = f / ksteps, k = f - n * ksteps;
    int col = (n << 4) + (lane & 15);
    int kk  = (k << 5) + ((lane >> 4) << 4) + (v << 1);
    float a0 = (kk     < Kreal) ? W[(size_t)kk * Ncols + col]       : 0.0f;
    float a1 = (kk + 1 < Kreal) ? W[(size_t)(kk + 1) * Ncols + col] : 0.0f;
    dst[i] = pack2h(a0, a1);
  }
}

// A-fragment: 16x32 f16 tile from LDS (row-major, stride kpad halves).
// Lane L: row m = L%16; K contiguous {kh..kh+15} with kh=(L/16)*16.
__device__ __forceinline__ v16h load_a(const _Float16* base, int kstep, int lane, int kpad) {
  int m = lane & 15, kh = (lane >> 4) << 4;
  const _Float16* p = base + m * kpad + (kstep << 5) + kh;
  v16h a;
  *(v8h*)&a         = *(const v8h*)p;
  *(((v8h*)&a) + 1) = *(const v8h*)(p + 8);
  return a;
}
__device__ __forceinline__ v16h load_b(const unsigned* wbuf, int frag, int lane) {
  return *(const v16h*)(wbuf + frag * 256 + lane * 8);
}

#define WMMA_F16(A, B, C) \
  __builtin_amdgcn_wmma_f32_16x16x32_f16(false, (A), false, (B), (short)0, (C), false, false)

__global__ void degree_kernel(const int* __restrict__ ei, float* __restrict__ cnt, int E) {
  int i = blockIdx.x * blockDim.x + threadIdx.x;
  if (i < E) atomicAdd(&cnt[ei[i]], 1.0f);
}

__global__ __launch_bounds__(256) void egcl_edge_kernel(
    const float* __restrict__ h, const float* __restrict__ x,
    const int* __restrict__ ei, const float* __restrict__ ea,
    const float* __restrict__ w_e1, const float* __restrict__ b_e1,
    const float* __restrict__ w_e2, const float* __restrict__ b_e2,
    const float* __restrict__ w_c1, const float* __restrict__ b_c1,
    const float* __restrict__ w_c2,
    const float* __restrict__ self_mult, const float* __restrict__ other_mult,
    float* __restrict__ agg, float* __restrict__ x_acc,
    int Bn, int Nn, int En)
{
  __shared__ unsigned we1f[KE_STEPS * 4 * 256];   // 20 KB
  __shared__ unsigned we2f[2 * 4 * 256];          //  8 KB
  __shared__ unsigned wc1f[2 * 4 * 256];          //  8 KB
  __shared__ _Float16 sA[8][16 * KPAD_E];         // 40 KB : per-wave m_in tiles
  __shared__ _Float16 sM[8][16 * MSG];            // 16 KB : per-wave m1/m2 staging
  __shared__ int      sIdx[8][16];
  __shared__ float    sPhi[8][16];

  const int tid = threadIdx.x, nthr = blockDim.x;
  fill_wfrags(w_e1, 2 * F_IN + 1 + EDGE_F, KE_STEPS, MSG, we1f, tid, nthr);
  fill_wfrags(w_e2, MSG, 2, MSG, we2f, tid, nthr);
  fill_wfrags(w_c1, MSG, 2, MSG, wc1f, tid, nthr);
  __syncthreads();

  const int lane = tid & 31, wv = tid >> 5;
  const int ET = (En + 15) >> 4;
  const long long gw = (long long)blockIdx.x * 8 + wv;
  if (gw >= (long long)Bn * ET) return;
  const int b  = (int)(gw / ET);
  const int e0 = (int)(gw % ET) << 4;
  const int em = lane & 15, half = lane >> 4;
  const int e  = e0 + em;
  const bool ev = e < En;
  const int ec = ev ? e : 0;
  const int r = ei[ec], c = ei[En + ec];
  if (!half) sIdx[wv][em] = r;

  // ---- stage m_in = [h_row | h_col | radial | edge_attr | 0-pad] as f16 ----
  const float* hb = h + (size_t)b * Nn * F_IN;
  _Float16* mrow = &sA[wv][em * KPAD_E];
  {
    const float4* s0 = (const float4*)(hb + (size_t)r * F_IN + half * 32);
    const float4* s1 = (const float4*)(hb + (size_t)c * F_IN + half * 32);
    for (int j = 0; j < 8; ++j) {
      float4 f = s0[j]; int o = half * 32 + j * 4;
      mrow[o] = (_Float16)f.x; mrow[o+1] = (_Float16)f.y;
      mrow[o+2] = (_Float16)f.z; mrow[o+3] = (_Float16)f.w;
      f = s1[j]; o += 64;
      mrow[o] = (_Float16)f.x; mrow[o+1] = (_Float16)f.y;
      mrow[o+2] = (_Float16)f.z; mrow[o+3] = (_Float16)f.w;
    }
  }
  float4 xr = {0,0,0,0}, xc = {0,0,0,0};
  if (!half) {
    const float4* xb = (const float4*)(x + (size_t)b * Nn * 4);
    xr = xb[r]; xc = xb[c];
    float dx = xr.x - xc.x, dy = xr.y - xc.y, dz = xr.z - xc.z, dw = xr.w - xc.w;
    mrow[2 * F_IN] = (_Float16)(dy*dy + dz*dz + dw*dw - dx*dx);  // (-,+,+,+)
    const float4* ar = (const float4*)(ea + ((size_t)b * En + ec) * EDGE_F);
    for (int j = 0; j < 4; ++j) {
      float4 f = ar[j]; int o = 2 * F_IN + 1 + j * 4;
      mrow[o] = (_Float16)f.x; mrow[o+1] = (_Float16)f.y;
      mrow[o+2] = (_Float16)f.z; mrow[o+3] = (_Float16)f.w;
    }
  } else {
    for (int k = 2 * F_IN + 1 + EDGE_F; k < KPAD_E; ++k) mrow[k] = (_Float16)0.0f;
  }
  asm volatile("s_wait_dscnt 0" ::: "memory");

  const int ncol = lane & 15, mrow0 = (lane >> 4) << 3;

  // ---- GEMM1: m1 = silu(m_in @ W_e1 + b_e1)  [16 x 64] ----
  v8f a0 = {}, a1 = {}, a2 = {}, a3 = {};
  for (int k = 0; k < KE_STEPS; ++k) {
    v16h a = load_a(&sA[wv][0], k, lane, KPAD_E);
    a0 = WMMA_F16(a, load_b(we1f, 0 * KE_STEPS + k, lane), a0);
    a1 = WMMA_F16(a, load_b(we1f, 1 * KE_STEPS + k, lane), a1);
    a2 = WMMA_F16(a, load_b(we1f, 2 * KE_STEPS + k, lane), a2);
    a3 = WMMA_F16(a, load_b(we1f, 3 * KE_STEPS + k, lane), a3);
  }
  {
    v8f cc[4] = {a0, a1, a2, a3};
    for (int j = 0; j < 4; ++j) {
      float bias = b_e1[j * 16 + ncol];
      for (int v = 0; v < 8; ++v)
        sM[wv][(mrow0 + v) * MSG + j * 16 + ncol] = (_Float16)silu_f(cc[j][v] + bias);
    }
  }
  asm volatile("s_wait_dscnt 0" ::: "memory");

  // ---- GEMM2: m2 = softsign(m1 @ W_e2 + b_e2); scatter-add to agg ----
  v16h q0 = load_a(&sM[wv][0], 0, lane, MSG);
  v16h q1 = load_a(&sM[wv][0], 1, lane, MSG);
  v8f d0 = {}, d1 = {}, d2 = {}, d3 = {};
  d0 = WMMA_F16(q0, load_b(we2f, 0, lane), d0); d0 = WMMA_F16(q1, load_b(we2f, 1, lane), d0);
  d1 = WMMA_F16(q0, load_b(we2f, 2, lane), d1); d1 = WMMA_F16(q1, load_b(we2f, 3, lane), d1);
  d2 = WMMA_F16(q0, load_b(we2f, 4, lane), d2); d2 = WMMA_F16(q1, load_b(we2f, 5, lane), d2);
  d3 = WMMA_F16(q0, load_b(we2f, 6, lane), d3); d3 = WMMA_F16(q1, load_b(we2f, 7, lane), d3);
  {
    v8f dd[4] = {d0, d1, d2, d3};
    for (int j = 0; j < 4; ++j) {
      int n = j * 16 + ncol;
      float bias = b_e2[n];
      for (int v = 0; v < 8; ++v) {
        int M = mrow0 + v;
        float m2 = ssign_f(dd[j][v] + bias);
        sM[wv][M * MSG + n] = (_Float16)m2;
        if (e0 + M < En)
          atomicAdd(&agg[((size_t)b * Nn + sIdx[wv][M]) * MSG + n], m2);
      }
    }
  }
  asm volatile("s_wait_dscnt 0" ::: "memory");

  // ---- GEMM3 + projection: phi = silu(m2 @ W_c1 + b_c1) @ w_c2  [16 x 1] ----
  v16h p0 = load_a(&sM[wv][0], 0, lane, MSG);
  v16h p1 = load_a(&sM[wv][0], 1, lane, MSG);
  v8f g0 = {}, g1 = {}, g2 = {}, g3 = {};
  g0 = WMMA_F16(p0, load_b(wc1f, 0, lane), g0); g0 = WMMA_F16(p1, load_b(wc1f, 1, lane), g0);
  g1 = WMMA_F16(p0, load_b(wc1f, 2, lane), g1); g1 = WMMA_F16(p1, load_b(wc1f, 3, lane), g1);
  g2 = WMMA_F16(p0, load_b(wc1f, 4, lane), g2); g2 = WMMA_F16(p1, load_b(wc1f, 5, lane), g2);
  g3 = WMMA_F16(p0, load_b(wc1f, 6, lane), g3); g3 = WMMA_F16(p1, load_b(wc1f, 7, lane), g3);
  float acc[8] = {0,0,0,0,0,0,0,0};
  {
    v8f gg[4] = {g0, g1, g2, g3};
    for (int j = 0; j < 4; ++j) {
      int n = j * 16 + ncol;
      float bias = b_c1[n], wv2 = w_c2[n];
      for (int v = 0; v < 8; ++v) acc[v] += silu_f(gg[j][v] + bias) * wv2;
    }
  }
  for (int m = 1; m < 16; m <<= 1)
    for (int v = 0; v < 8; ++v) acc[v] += __shfl_xor(acc[v], m, 32);
  if (lane == 0)  for (int v = 0; v < 8; ++v) sPhi[wv][v]     = acc[v];
  if (lane == 16) for (int v = 0; v < 8; ++v) sPhi[wv][8 + v] = acc[v];
  asm volatile("s_wait_dscnt 0" ::: "memory");

  // ---- coordinate scatter: (sm*x_r + om*x_c) * phi into x_acc ----
  if (!half && ev) {
    float ph = sPhi[wv][em];
    float sm = self_mult[0], om = other_mult[0];
    float* dst = &x_acc[((size_t)b * Nn + r) * 4];
    atomicAdd(&dst[0], (sm * xr.x + om * xc.x) * ph);
    atomicAdd(&dst[1], (sm * xr.y + om * xc.y) * ph);
    atomicAdd(&dst[2], (sm * xr.z + om * xc.z) * ph);
    atomicAdd(&dst[3], (sm * xr.w + om * xc.w) * ph);
  }
}

__global__ __launch_bounds__(256) void egcl_node_kernel(
    const float* __restrict__ h, const float* __restrict__ x,
    const float* __restrict__ w_f1, const float* __restrict__ b_f1,
    const float* __restrict__ w_f2, const float* __restrict__ b_f2,
    const float* __restrict__ agg, const float* __restrict__ x_acc,
    const float* __restrict__ cnt,
    float* __restrict__ out_h, float* __restrict__ out_x, int Bn, int Nn)
{
  __shared__ unsigned wf1f[KN_STEPS * 4 * 256];   // 16 KB
  __shared__ unsigned wf2f[2 * 4 * 256];          //  8 KB
  __shared__ _Float16 sA[8][16 * KN];             // 32 KB
  __shared__ _Float16 sM[8][16 * MSG];            // 16 KB

  const int tid = threadIdx.x, nthr = blockDim.x;
  fill_wfrags(w_f1, KN, KN_STEPS, MSG, wf1f, tid, nthr);
  fill_wfrags(w_f2, MSG, 2, MSG, wf2f, tid, nthr);
  __syncthreads();

  const int lane = tid & 31, wv = tid >> 5;
  const long long rows = (long long)Bn * Nn;
  const long long tiles = (rows + 15) >> 4;
  const long long gw = (long long)blockIdx.x * 8 + wv;
  if (gw >= tiles) return;
  const long long g0 = gw << 4;
  const int em = lane & 15, half = lane >> 4;
  const long long g = g0 + em;
  const bool gvld = g < rows;
  const long long gc = gvld ? g : 0;

  // stage [h | agg] as f16: half0 -> cols 0..63 (h), half1 -> cols 64..127 (agg)
  {
    const float* src = half ? (agg + gc * MSG) : (h + gc * F_IN);
    _Float16* dst = &sA[wv][em * KN + half * 64];
    const float4* s4 = (const float4*)src;
    for (int j = 0; j < 16; ++j) {
      float4 f = s4[j]; int o = j * 4;
      dst[o] = (_Float16)f.x; dst[o+1] = (_Float16)f.y;
      dst[o+2] = (_Float16)f.z; dst[o+3] = (_Float16)f.w;
    }
  }
  asm volatile("s_wait_dscnt 0" ::: "memory");

  const int ncol = lane & 15, mrow0 = (lane >> 4) << 3;

  // GEMM1: silu([h|agg] @ W_f1 + b_f1)
  v8f a0 = {}, a1 = {}, a2 = {}, a3 = {};
  for (int k = 0; k < KN_STEPS; ++k) {
    v16h a = load_a(&sA[wv][0], k, lane, KN);
    a0 = WMMA_F16(a, load_b(wf1f, 0 * KN_STEPS + k, lane), a0);
    a1 = WMMA_F16(a, load_b(wf1f, 1 * KN_STEPS + k, lane), a1);
    a2 = WMMA_F16(a, load_b(wf1f, 2 * KN_STEPS + k, lane), a2);
    a3 = WMMA_F16(a, load_b(wf1f, 3 * KN_STEPS + k, lane), a3);
  }
  {
    v8f cc[4] = {a0, a1, a2, a3};
    for (int j = 0; j < 4; ++j) {
      float bias = b_f1[j * 16 + ncol];
      for (int v = 0; v < 8; ++v)
        sM[wv][(mrow0 + v) * MSG + j * 16 + ncol] = (_Float16)silu_f(cc[j][v] + bias);
    }
  }
  asm volatile("s_wait_dscnt 0" ::: "memory");

  // GEMM2: h_out = softsign(m @ W_f2 + b_f2)
  v16h q0 = load_a(&sM[wv][0], 0, lane, MSG);
  v16h q1 = load_a(&sM[wv][0], 1, lane, MSG);
  v8f d0 = {}, d1 = {}, d2 = {}, d3 = {};
  d0 = WMMA_F16(q0, load_b(wf2f, 0, lane), d0); d0 = WMMA_F16(q1, load_b(wf2f, 1, lane), d0);
  d1 = WMMA_F16(q0, load_b(wf2f, 2, lane), d1); d1 = WMMA_F16(q1, load_b(wf2f, 3, lane), d1);
  d2 = WMMA_F16(q0, load_b(wf2f, 4, lane), d2); d2 = WMMA_F16(q1, load_b(wf2f, 5, lane), d2);
  d3 = WMMA_F16(q0, load_b(wf2f, 6, lane), d3); d3 = WMMA_F16(q1, load_b(wf2f, 7, lane), d3);
  {
    v8f dd[4] = {d0, d1, d2, d3};
    for (int j = 0; j < 4; ++j) {
      int n = j * 16 + ncol;
      float bias = b_f2[n];
      for (int v = 0; v < 8; ++v) {
        long long gg = g0 + mrow0 + v;
        if (gg < rows) out_h[(size_t)gg * F_OUT + n] = ssign_f(dd[j][v] + bias);
      }
    }
  }

  // x_updated = x + x_acc / max(count,1)
  if (!half && gvld) {
    int node = (int)(gc % Nn);
    float cv = cnt[node]; if (cv < 1.0f) cv = 1.0f;
    const float* xs = x + gc * 4;
    const float* as = x_acc + gc * 4;
    float* xo = out_x + gc * 4;
    for (int dck = 0; dck < 4; ++dck) xo[dck] = xs[dck] + as[dck] / cv;
  }
}

extern "C" void kernel_launch(void* const* d_in, const int* in_sizes, int n_in,
                              void* d_out, int out_size, void* d_ws, size_t ws_size,
                              hipStream_t stream) {
  const float* h    = (const float*)d_in[0];
  const float* x    = (const float*)d_in[1];
  const int*   ei   = (const int*)  d_in[2];
  const float* ea   = (const float*)d_in[3];
  const float* w_e1 = (const float*)d_in[4];
  const float* b_e1 = (const float*)d_in[5];
  const float* w_e2 = (const float*)d_in[6];
  const float* b_e2 = (const float*)d_in[7];
  const float* w_f1 = (const float*)d_in[8];
  const float* b_f1 = (const float*)d_in[9];
  const float* w_f2 = (const float*)d_in[10];
  const float* b_f2 = (const float*)d_in[11];
  const float* w_c1 = (const float*)d_in[12];
  const float* b_c1 = (const float*)d_in[13];
  const float* w_c2 = (const float*)d_in[14];
  const float* smul = (const float*)d_in[15];
  const float* omul = (const float*)d_in[16];

  const int E = in_sizes[2] / 2;
  const int B = (int)((long long)in_sizes[3] / ((long long)E * EDGE_F));
  const long long BN = (long long)in_sizes[1] / 4;
  const int N = (int)(BN / B);

  float* agg  = (float*)d_ws;
  float* xacc = agg + (size_t)B * N * MSG;
  float* cntv = xacc + (size_t)B * N * 4;
  const size_t zbytes = ((size_t)B * N * MSG + (size_t)B * N * 4 + (size_t)N) * sizeof(float);
  (void)hipMemsetAsync(d_ws, 0, zbytes, stream);

  degree_kernel<<<(E + 255) / 256, 256, 0, stream>>>(ei, cntv, E);

  const long long ewaves = (long long)B * ((E + 15) / 16);
  egcl_edge_kernel<<<(int)((ewaves + 7) / 8), 256, 0, stream>>>(
      h, x, ei, ea, w_e1, b_e1, w_e2, b_e2, w_c1, b_c1, w_c2,
      smul, omul, agg, xacc, B, N, E);

  float* out_h = (float*)d_out;
  float* out_x = out_h + (size_t)B * N * MSG;
  const long long ntiles = (BN + 15) / 16;
  egcl_node_kernel<<<(int)((ntiles + 7) / 8), 256, 0, stream>>>(
      h, x, w_f1, b_f1, w_f2, b_f2, agg, xacc, cntv, out_h, out_x, B, N);
}
